// MultiHeadAttention_21887153340754
// MI455X (gfx1250) — matrix operand order repaired and checked by reading, not compiled, not run on hardware
//
#include <hip/hip_runtime.h>

// ---------------------------------------------------------------------------
// MI455X (gfx1250) wave32 WMMA implementation of the MLP->attention pipeline.
//   All matmuls: v_wmma_f32_16x16x32_f16 (f16 in, f32 accum).
//   GEMM: 128x64 block tile, 32x64 per wave (2 A-frags x 4 B-frags = 8 WMMA
//         per K-step, B fragments reused across the two row strips).
//   Attention fused flash-style (linear row-sum normalization -> one pass).
// ---------------------------------------------------------------------------

typedef __attribute__((ext_vector_type(16))) _Float16 v16h;
typedef __attribute__((ext_vector_type(8)))  float    v8f;

union FragU {      // 8 VGPRs: element pair (2r,2r+1) lives in VGPR r
  v16h  v;
  uint4 q[2];
};

__device__ __forceinline__ v8f wmma_f16(const FragU& a, const FragU& b, v8f c) {
  return __builtin_amdgcn_wmma_f32_16x16x32_f16(
      /*neg_a=*/false, a.v, /*neg_b=*/false, b.v,
      /*c_mod=*/(short)0, c, /*reuse_a=*/false, /*reuse_b=*/false);
}

// f32 [R,C] -> f16 [R,Cpad], zero padded
__global__ void cvt_pad_f16(const float* __restrict__ X, _Float16* __restrict__ Y,
                            int R, int C, int Cpad) {
  int idx = blockIdx.x * blockDim.x + threadIdx.x;
  if (idx >= R * Cpad) return;
  int r = idx / Cpad, c = idx - r * Cpad;
  Y[idx] = (c < C) ? (_Float16)X[r * C + c] : (_Float16)0.f;
}

// f32 W[K,N] -> f16 Wt[Npad,Kpad] (transposed + zero padded)
__global__ void cvt_wt_f16(const float* __restrict__ W, _Float16* __restrict__ Wt,
                           int K, int N, int Kpad, int Npad) {
  int idx = blockIdx.x * blockDim.x + threadIdx.x;
  if (idx >= Npad * Kpad) return;
  int n = idx / Kpad, k = idx - n * Kpad;
  Wt[idx] = (k < K && n < N) ? (_Float16)W[(size_t)k * N + n] : (_Float16)0.f;
}

// C[M,N] = act(A[M,K] * Wt[N,K]^T + bias), block tile 128x64, wave tile 32x64.
// MODE 0: f16 row-major; MODE 1: f16 transposed (C[n*ldc+m]); MODE 2: f32, col<Nout
template<int ACT, int MODE>
__global__ __launch_bounds__(128)
void gemm_wmma(const _Float16* __restrict__ A, const _Float16* __restrict__ Wt,
               const float* __restrict__ bias, void* __restrict__ C,
               int K, int Nout, int ldc) {
  const int lane   = threadIdx.x & 31;
  const int wave   = threadIdx.x >> 5;
  const int lane16 = lane & 15;
  const int laneh  = lane >> 4;

  const int rowBase = blockIdx.y * 128 + wave * 32;
  const int colBase = blockIdx.x * 64;

  const _Float16* Arow0 = A + (size_t)(rowBase + lane16) * K;
  const _Float16* Arow1 = A + (size_t)(rowBase + 16 + lane16) * K;

  v8f acc[2][4] = {};

  for (int kb = 0; kb < K; kb += 32) {
    FragU a0, a1;          // A 16x32: two 16B segments per lane
    a0.q[0] = *(const uint4*)(Arow0 + kb + 8 * laneh);
    a0.q[1] = *(const uint4*)(Arow0 + kb + 16 + 8 * laneh);
    a1.q[0] = *(const uint4*)(Arow1 + kb + 8 * laneh);
    a1.q[1] = *(const uint4*)(Arow1 + kb + 16 + 8 * laneh);
    __builtin_prefetch(Arow0 + kb + 256, 0, 0);   // speculative, next K blocks
    __builtin_prefetch(Arow1 + kb + 256, 0, 0);
#pragma unroll
    for (int j = 0; j < 4; ++j) {
      const _Float16* Wrow = Wt + (size_t)(colBase + 16 * j + lane16) * K + kb + 8 * laneh;
      FragU b;             // B 32x16: two 16B segments per lane, reused for both A frags
      b.q[0] = *(const uint4*)(Wrow);
      b.q[1] = *(const uint4*)(Wrow + 16);
      acc[0][j] = wmma_f16(a0, b, acc[0][j]);
      acc[1][j] = wmma_f16(a1, b, acc[1][j]);
    }
  }

#pragma unroll
  for (int j = 0; j < 4; ++j) {
    int col = colBase + 16 * j + lane16;
    float bv = (col < Nout) ? bias[col] : 0.f;
#pragma unroll
    for (int i = 0; i < 2; ++i) {
#pragma unroll
      for (int r = 0; r < 8; ++r) {
        int   orow = rowBase + 16 * i + r + 8 * laneh;  // C layout: M=r+8*laneh, N=lane16
        float v    = acc[i][j][r] + bv;
        if (ACT) v = v > 0.f ? v : 0.f;
        if (MODE == 0)      ((_Float16*)C)[(size_t)orow * ldc + col] = (_Float16)v;
        else if (MODE == 1) ((_Float16*)C)[(size_t)col * ldc + orow] = (_Float16)v;
        else if (col < Nout) ((float*)C)[(size_t)orow * ldc + col]   = v;
      }
    }
  }
}

// Fused per-head attention: one pass over keys/values, accumulating
// numerator (P@V) via a second WMMA and the denominator (row sums) in VALU.
__global__ __launch_bounds__(128)
void attn_wmma(const _Float16* __restrict__ Q3, const _Float16* __restrict__ K3,
               const _Float16* __restrict__ VT, _Float16* __restrict__ ATT) {
  __shared__ __align__(16) _Float16 lds[4][16][32];   // per-wave P staging
  const int lane   = threadIdx.x & 31;
  const int wave   = threadIdx.x >> 5;
  const int lane16 = lane & 15;
  const int laneh  = lane >> 4;
  const int rowBase = blockIdx.x * 64 + wave * 16;    // 16 query rows per wave
  const int hoff    = blockIdx.y * 64;                // head column offset

  // Q fragments (depth 64 -> two K=32 steps), relu already applied upstream
  const _Float16* Qrow = Q3 + (size_t)(rowBase + lane16) * 512 + hoff;
  FragU qf[2];
#pragma unroll
  for (int s = 0; s < 2; ++s) {
    qf[s].q[0] = *(const uint4*)(Qrow + 32 * s + 8 * laneh);
    qf[s].q[1] = *(const uint4*)(Qrow + 32 * s + 16 + 8 * laneh);
  }

  v8f   oacc[4] = {};
  float dsum[8] = {0.f, 0.f, 0.f, 0.f, 0.f, 0.f, 0.f, 0.f};

  for (int m0 = 0; m0 < 4096; m0 += 32) {
    v8f s0 = {}, s1 = {};
#pragma unroll
    for (int s = 0; s < 2; ++s) {       // S = Q @ K^T : B operand = rows of K3
      const _Float16* Kr0 = K3 + (size_t)(m0 + lane16) * 512 + hoff + 32 * s + 8 * laneh;
      FragU b0; b0.q[0] = *(const uint4*)(Kr0); b0.q[1] = *(const uint4*)(Kr0 + 16);
      s0 = wmma_f16(qf[s], b0, s0);
      const _Float16* Kr1 = K3 + (size_t)(m0 + 16 + lane16) * 512 + hoff + 32 * s + 8 * laneh;
      FragU b1; b1.q[0] = *(const uint4*)(Kr1); b1.q[1] = *(const uint4*)(Kr1 + 16);
      s1 = wmma_f16(qf[s], b1, s1);
    }
    // scale by 1/sqrt(depth), accumulate denominator, stage P tile (16x32) in LDS
#pragma unroll
    for (int r = 0; r < 8; ++r) {
      float a = s0[r] * 0.125f;
      float b = s1[r] * 0.125f;
      dsum[r] += a + b;
      lds[wave][r + 8 * laneh][lane16]      = (_Float16)a;
      lds[wave][r + 8 * laneh][16 + lane16] = (_Float16)b;
    }
    __syncthreads();
    FragU p;                             // re-read P in WMMA A layout (ds_load_b128)
    const _Float16* prow = &lds[wave][lane16][0];
    p.q[0] = *(const uint4*)(prow + 8 * laneh);
    p.q[1] = *(const uint4*)(prow + 16 + 8 * laneh);
#pragma unroll
    for (int j = 0; j < 4; ++j) {        // att += P @ V ; V pre-transposed -> b128 loads
      const _Float16* Vrow = VT + (size_t)(hoff + 16 * j + lane16) * 4096 + m0 + 8 * laneh;
      FragU vb; vb.q[0] = *(const uint4*)(Vrow); vb.q[1] = *(const uint4*)(Vrow + 16);
      oacc[j] = wmma_f16(p, vb, oacc[j]);
    }
    __syncthreads();
  }

  // row-sum reduction across the 16 lanes of each half (wave32 shuffles)
  float inv[8];
#pragma unroll
  for (int r = 0; r < 8; ++r) {
    float d = dsum[r];
    d += __shfl_xor(d, 1, 32);
    d += __shfl_xor(d, 2, 32);
    d += __shfl_xor(d, 4, 32);
    d += __shfl_xor(d, 8, 32);
    inv[r] = 1.f / (d + 1e-8f);
  }
#pragma unroll
  for (int j = 0; j < 4; ++j) {
    int col = hoff + 16 * j + lane16;
#pragma unroll
    for (int r = 0; r < 8; ++r) {
      int orow = rowBase + r + 8 * laneh;
      ATT[(size_t)orow * 512 + col] = (_Float16)(oacc[j][r] * inv[r]);
    }
  }
}

extern "C" void kernel_launch(void* const* d_in, const int* in_sizes, int n_in,
                              void* d_out, int out_size, void* d_ws, size_t ws_size,
                              hipStream_t stream) {
  (void)in_sizes; (void)n_in; (void)out_size; (void)ws_size;
  const float* batch = (const float*)d_in[0];
  const float* wo_w  = (const float*)d_in[19];
  const float* wo_b  = (const float*)d_in[20];

  char* ws = (char*)d_ws;                               // 52 MB total
  _Float16* Xb  = (_Float16*)(ws);                      // [4096,1024]   8 MB
  _Float16* H1  = (_Float16*)(ws +  8388608);           // [4096,2048]  16 MB
  _Float16* H2  = (_Float16*)(ws + 25165824);           // [4096,1024]   8 MB
  _Float16* Q3  = (_Float16*)(ws + 33554432);           // [4096,512]    4 MB
  _Float16* K3s = (_Float16*)(ws + 37748736);           // [4096,512]    4 MB
  _Float16* VT  = (_Float16*)(ws + 41943040);           // [512,4096]    4 MB (V^T)
  _Float16* ATT = (_Float16*)(ws + 46137344);           // [4096,512]    4 MB
  _Float16* Wt  = (_Float16*)(ws + 50331648);           // scratch <= 4 MB

  cvt_pad_f16<<<(4096 * 1024 + 255) / 256, 256, 0, stream>>>(batch, Xb, 4096, 1000, 1024);

  const int bases[3] = {1, 7, 13};
  _Float16* q3k3[2] = {Q3, K3s};
  for (int p = 0; p < 3; ++p) {
    const float* w1 = (const float*)d_in[bases[p] + 0];
    const float* b1 = (const float*)d_in[bases[p] + 1];
    const float* w2 = (const float*)d_in[bases[p] + 2];
    const float* b2 = (const float*)d_in[bases[p] + 3];
    const float* w3 = (const float*)d_in[bases[p] + 4];
    const float* b3 = (const float*)d_in[bases[p] + 5];

    cvt_wt_f16<<<(2048 * 1024 + 255) / 256, 256, 0, stream>>>(w1, Wt, 1000, 2048, 1024, 2048);
    gemm_wmma<1, 0><<<dim3(32, 32), 128, 0, stream>>>(Xb, Wt, b1, H1, 1024, 2048, 2048);

    cvt_wt_f16<<<(1024 * 2048 + 255) / 256, 256, 0, stream>>>(w2, Wt, 2048, 1024, 2048, 1024);
    gemm_wmma<1, 0><<<dim3(16, 32), 128, 0, stream>>>(H1, Wt, b2, H2, 2048, 1024, 1024);

    cvt_wt_f16<<<(512 * 1024 + 255) / 256, 256, 0, stream>>>(w3, Wt, 1024, 512, 1024, 512);
    if (p < 2)  // relu here == relu(split_heads(.)) since relu is elementwise
      gemm_wmma<1, 0><<<dim3(8, 32), 128, 0, stream>>>(H2, Wt, b3, q3k3[p], 1024, 512, 512);
    else        // V: no relu; store transposed so attention B-fragments are contiguous
      gemm_wmma<0, 1><<<dim3(8, 32), 128, 0, stream>>>(H2, Wt, b3, VT, 1024, 512, 4096);
  }

  attn_wmma<<<dim3(64, 8), 128, 0, stream>>>(Q3, K3s, VT, ATT);

  cvt_wt_f16<<<(1024 * 512 + 255) / 256, 256, 0, stream>>>(wo_w, Wt, 512, 1000, 512, 1024);
  gemm_wmma<0, 2><<<dim3(16, 32), 128, 0, stream>>>(ATT, Wt, wo_b, d_out, 512, 1000, 1000);
}
